// FullAttention_52415780880838
// MI455X (gfx1250) — matrix operand order repaired and checked by reading, not compiled, not run on hardware
//
#include <hip/hip_runtime.h>
#include <hip/hip_bf16.h>

typedef __attribute__((ext_vector_type(16))) _Float16 v16h;
typedef __attribute__((ext_vector_type(8)))  float    v8f;

// D = A(16x32 f16) * B(32x16 f16) + C(16x16 f32)
__device__ __forceinline__ v8f wmma_f16(v16h a, v16h b, v8f c) {
  return __builtin_amdgcn_wmma_f32_16x16x32_f16(false, a, false, b, (short)0, c,
                                                false, false);
}

// SWAPX16 within wave32: group-of-32 swizzle, xor=0x10, and=0x1f -> 0x401F
__device__ __forceinline__ float lane_xor16(float v) {
  return __int_as_float(__builtin_amdgcn_ds_swizzle(__float_as_int(v), 0x401F));
}
__device__ __forceinline__ float lane_gather(float v, int srclane) {
  return __int_as_float(
      __builtin_amdgcn_ds_bpermute(srclane << 2, __float_as_int(v)));
}

// A-matrix 16x32 f16 fragment: halves 0..7 -> k = kb+0..7, halves 8..15 -> k = kb+16..23
// (kb = 8 for lanes >= 16). `row` points at element k=0 of this lane's matrix row.
__device__ __forceinline__ v16h load_fragA(const float* __restrict__ row, int kb) {
  float4 f0 = *reinterpret_cast<const float4*>(row + kb);
  float4 f1 = *reinterpret_cast<const float4*>(row + kb + 4);
  float4 f2 = *reinterpret_cast<const float4*>(row + kb + 16);
  float4 f3 = *reinterpret_cast<const float4*>(row + kb + 20);
  v16h r;
  r[0]  = (_Float16)f0.x; r[1]  = (_Float16)f0.y; r[2]  = (_Float16)f0.z; r[3]  = (_Float16)f0.w;
  r[4]  = (_Float16)f1.x; r[5]  = (_Float16)f1.y; r[6]  = (_Float16)f1.z; r[7]  = (_Float16)f1.w;
  r[8]  = (_Float16)f2.x; r[9]  = (_Float16)f2.y; r[10] = (_Float16)f2.z; r[11] = (_Float16)f2.w;
  r[12] = (_Float16)f3.x; r[13] = (_Float16)f3.y; r[14] = (_Float16)f3.z; r[15] = (_Float16)f3.w;
  return r;
}

// B-matrix 32x16 f16 fragment: halves 0..7 -> k = kb+0..7, halves 8..15 -> k = kb+16..23 (kb = 8 for lanes >= 16).
// `p` points at element (k = kb) of this lane's matrix column, contiguous in memory.
__device__ __forceinline__ v16h load_fragB(const float* __restrict__ p) {
  float4 f0 = *reinterpret_cast<const float4*>(p);
  float4 f1 = *reinterpret_cast<const float4*>(p + 4);
  float4 f2 = *reinterpret_cast<const float4*>(p + 16);
  float4 f3 = *reinterpret_cast<const float4*>(p + 20);
  v16h r;
  r[0]  = (_Float16)f0.x; r[1]  = (_Float16)f0.y; r[2]  = (_Float16)f0.z; r[3]  = (_Float16)f0.w;
  r[4]  = (_Float16)f1.x; r[5]  = (_Float16)f1.y; r[6]  = (_Float16)f1.z; r[7]  = (_Float16)f1.w;
  r[8]  = (_Float16)f2.x; r[9]  = (_Float16)f2.y; r[10] = (_Float16)f2.z; r[11] = (_Float16)f2.w;
  r[12] = (_Float16)f3.x; r[13] = (_Float16)f3.y; r[14] = (_Float16)f3.z; r[15] = (_Float16)f3.w;
  return r;
}

// B fragment from V with row stride (k runs over s-rows, column d fixed per lane).
__device__ __forceinline__ v16h load_fragV(const float* __restrict__ col, int stride) {
  v16h r;
#pragma unroll
  for (int i = 0; i < 16; ++i) r[i] = (_Float16)col[(size_t)(i < 8 ? i : i + 8) * stride];
  return r;
}

// B=4, N=32, L=S=256, H=8, E=D=64.  One wave per (b,n,h, 32-row L tile):
// two 16-row subtiles share each K/V fragment -> half the vmem instructions.
__global__ __launch_bounds__(256) void fa_fwd_kernel(
    const float* __restrict__ Q, const float* __restrict__ K,
    const float* __restrict__ V, float* __restrict__ O) {
  constexpr int Nn = 32, L = 256, S = 256, H = 8, E = 64, Dd = 64;
  constexpr int HE = H * E;  // 512: l/s stride in Q,K
  constexpr int HD = H * Dd; // 512: l/s stride in V,O

  const int lane  = threadIdx.x & 31;
  const int wave  = threadIdx.x >> 5;
  const int wid   = blockIdx.x * 8 + wave;  // 0 .. 8191
  const int lt32  = wid & 7;                // L/32 tiles
  const int h     = (wid >> 3) & 7;
  const int n     = (wid >> 6) & 31;
  const int b     = wid >> 11;

  const int half16 = (lane >> 4) & 1;
  const int lcol   = lane & 15;
  const int kb     = half16 * 8;    // A-fragment k base
  const int koff   = half16 * 16;   // unused; B fragments also use kb

  const size_t bn = (size_t)(b * Nn + n);
  const float* Qh = Q + bn * (size_t)L * HE + (size_t)h * E;
  const float* Kh = K + bn * (size_t)S * HE + (size_t)h * E;
  const float* Vh = V + bn * (size_t)S * HD + (size_t)h * Dd;
  float*       Oh = O + bn * (size_t)L * HD + (size_t)h * Dd;

  // Q as B-matrix (Q^T): lane column l, k runs over e. Two 16-row subtiles.
  v16h bQ[2][2];
#pragma unroll
  for (int lt = 0; lt < 2; ++lt) {
    const float* qrow = Qh + (size_t)(lt32 * 32 + lt * 16 + lcol) * HE + kb;
    bQ[lt][0] = load_fragB(qrow);
    bQ[lt][1] = load_fragB(qrow + 32);
  }

  v8f acc[2][4] = {{{}, {}, {}, {}}, {{}, {}, {}, {}}};
  float run_m[2] = {-3.0e38f, -3.0e38f};
  float run_s[2] = {0.0f, 0.0f};
  const float SC = 0.125f * 1.44269504088896f;  // (1/sqrt(E)) * log2(e)

  for (int s0 = 0; s0 < S; s0 += 32) {
    // ---- shared K fragments (A-matrix of scores^T) ----
    const float* krow0 = Kh + (size_t)(s0 + lcol) * HE;
    const float* krow1 = krow0 + (size_t)16 * HE;
    v16h a00 = load_fragA(krow0, kb);
    v16h a01 = load_fragA(krow0 + 32, kb);
    v16h a10 = load_fragA(krow1, kb);
    v16h a11 = load_fragA(krow1 + 32, kb);

    // ---- shared V fragments (B-matrix of P*V), 4 d-subtiles ----
    const float* vcol = Vh + (size_t)(s0 + kb) * HD + lcol;
    v16h bv0 = load_fragV(vcol, HD);
    v16h bv1 = load_fragV(vcol + 16, HD);
    v16h bv2 = load_fragV(vcol + 32, HD);
    v16h bv3 = load_fragV(vcol + 48, HD);

    if (s0 + 32 < S) {  // pull next chunk toward the caches while we compute
      __builtin_prefetch(krow0 + (size_t)32 * HE, 0, 0);
      __builtin_prefetch(vcol + (size_t)32 * HD, 0, 0);
    }

#pragma unroll
    for (int lt = 0; lt < 2; ++lt) {
      // ---- scores^T tile: D[s][l] = sum_e K[s][e] * Q[l][e] ----
      v8f sc0 = {}, sc1 = {};
      sc0 = wmma_f16(a00, bQ[lt][0], sc0);
      sc0 = wmma_f16(a01, bQ[lt][1], sc0);
      sc1 = wmma_f16(a10, bQ[lt][0], sc1);
      sc1 = wmma_f16(a11, bQ[lt][1], sc1);

      // ---- online softmax along s (per column l; split only across lane^16) ----
      float cmax = sc0[0];
#pragma unroll
      for (int j = 1; j < 8; ++j) cmax = fmaxf(cmax, sc0[j]);
#pragma unroll
      for (int j = 0; j < 8; ++j) cmax = fmaxf(cmax, sc1[j]);
      cmax = fmaxf(cmax, lane_xor16(cmax));

      const float newm = fmaxf(run_m[lt], cmax);
      const float corr = __builtin_amdgcn_exp2f((run_m[lt] - newm) * SC);
      run_m[lt] = newm;

      // P^T tile converts directly into the A-fragment of P*V (layouts coincide)
      v16h pf;
      float csum = 0.0f;
#pragma unroll
      for (int j = 0; j < 8; ++j) {
        float p = __builtin_amdgcn_exp2f((sc0[j] - newm) * SC);
        csum += p;
        pf[j] = (_Float16)p;
      }
#pragma unroll
      for (int j = 0; j < 8; ++j) {
        float p = __builtin_amdgcn_exp2f((sc1[j] - newm) * SC);
        csum += p;
        pf[8 + j] = (_Float16)p;
      }
      csum += lane_xor16(csum);
      run_s[lt] = run_s[lt] * corr + csum;

      // rescale accumulators: row l of acc register j is j + 8*half16
#pragma unroll
      for (int j = 0; j < 8; ++j) {
        float cj = lane_gather(corr, j + 8 * half16);
        acc[lt][0][j] *= cj; acc[lt][1][j] *= cj;
        acc[lt][2][j] *= cj; acc[lt][3][j] *= cj;
      }

      // ---- out += P * V ----
      acc[lt][0] = wmma_f16(pf, bv0, acc[lt][0]);
      acc[lt][1] = wmma_f16(pf, bv1, acc[lt][1]);
      acc[lt][2] = wmma_f16(pf, bv2, acc[lt][2]);
      acc[lt][3] = wmma_f16(pf, bv3, acc[lt][3]);
    }
  }

  // ---- finalize: divide by softmax sum, store [B,N,L,H,D] ----
#pragma unroll
  for (int lt = 0; lt < 2; ++lt) {
#pragma unroll
    for (int j = 0; j < 8; ++j) {
      float sj  = lane_gather(run_s[lt], j + 8 * half16);
      float inv = __builtin_amdgcn_rcpf(sj);
      const int l = lt32 * 32 + lt * 16 + j + 8 * half16;
      float* orow = Oh + (size_t)l * HD;
      orow[0  + lcol] = acc[lt][0][j] * inv;
      orow[16 + lcol] = acc[lt][1][j] * inv;
      orow[32 + lcol] = acc[lt][2][j] * inv;
      orow[48 + lcol] = acc[lt][3][j] * inv;
    }
  }
}

extern "C" void kernel_launch(void* const* d_in, const int* in_sizes, int n_in,
                              void* d_out, int out_size, void* d_ws, size_t ws_size,
                              hipStream_t stream) {
  (void)in_sizes; (void)n_in; (void)out_size; (void)d_ws; (void)ws_size;
  const float* Q = (const float*)d_in[0];
  const float* K = (const float*)d_in[1];
  const float* V = (const float*)d_in[2];
  float*       O = (float*)d_out;
  // 1024 heads * 8 L-tiles (32 rows each) = 8192 waves; 8 waves per block
  fa_fwd_kernel<<<1024, 256, 0, stream>>>(Q, K, V, O);
}
